// GCNEncoder_17463337025613
// MI455X (gfx1250) — matrix operand order repaired and checked by reading, not compiled, not run on hardware
//
#include <hip/hip_runtime.h>

// ---------------------------------------------------------------------------
// GCN encoder (3x [GCNConv -> ReLU -> BatchNorm]) for MI455X / gfx1250.
// GEMM runs on v_wmma_f32_16x16x32_bf16 (wave32 WMMA) with B fragments
// pre-swizzled into register layout (contiguous 32B loads per lane).
// Message passing uses L2-resident fp32 global atomics (h/agg are 25.6MB
// each; both fit in the 192MB L2, so the scatter is L2-atomic bound).
// ---------------------------------------------------------------------------

#define DF 64
static constexpr float EPSF = 1e-5f;

typedef __attribute__((ext_vector_type(16))) __bf16        v16bf;
typedef __attribute__((ext_vector_type(16))) unsigned short v16u;
typedef __attribute__((ext_vector_type(8)))  float          v8f;

__device__ __forceinline__ unsigned short f2bf_bits(float f) {
  union { float f; unsigned u; } a; a.f = f;
  unsigned u = a.u;
  u += 0x7FFFu + ((u >> 16) & 1u);          // round-to-nearest-even
  return (unsigned short)(u >> 16);
}

// ---------------- setup kernels ----------------

__global__ __launch_bounds__(256) void k_fill_deg(float* deg, int n) {
  int i = blockIdx.x * blockDim.x + threadIdx.x;
  if (i < n) deg[i] = 1.0f;                  // self-loop contribution
}

__global__ __launch_bounds__(256) void k_deg_accum(const long long* __restrict__ ei,
                                                   float* __restrict__ deg, long long e) {
  long long t = (long long)blockIdx.x * blockDim.x + threadIdx.x;
  if (t < e) atomicAdd(&deg[(int)ei[e + t]], 1.0f);   // dst = edge_index[1]
}

__global__ __launch_bounds__(256) void k_dinv(const float* __restrict__ deg,
                                              float* __restrict__ dinv, int n) {
  int i = blockIdx.x * blockDim.x + threadIdx.x;
  if (i < n) dinv[i] = rsqrtf(deg[i]);
}

// Convert W1/W2/W3 (fp32 64x64, row-major) to bf16 pre-swizzled into the
// WMMA B-fragment layout k_gemm_wmma uses (contiguous k per lane half):
//   out[((layer*8 + f)*32 + lane)*16 + j] = bf16( W[k, n] )
//   with f = ks*4 + nt, half = lane>>4, nlo = lane&15,
//        k = ks*32 + 16*half + j,  n = nt*16 + nlo.
// Each wave in the GEMM then loads fragment f as one contiguous 32B vector.
__global__ __launch_bounds__(256) void k_convert_w(const float* __restrict__ w1,
                                                   const float* __restrict__ w2,
                                                   const float* __restrict__ w3,
                                                   unsigned short* __restrict__ wswz) {
  int i = blockIdx.x * blockDim.x + threadIdx.x;
  if (i >= 3 * 4096) return;
  const int l    = i >> 12;          // layer
  const int r    = i & 4095;
  const int f    = r >> 9;           // fragment 0..7
  const int lane = (r >> 4) & 31;
  const int j    = r & 15;
  const int ks   = f >> 2;
  const int nt   = f & 3;
  const int half = lane >> 4;
  const int nlo  = lane & 15;
  const int k    = ks * 32 + 16 * half + j;
  const int n    = nt * 16 + nlo;
  const float* w = (l == 0) ? w1 : (l == 1) ? w2 : w3;
  wswz[i] = f2bf_bits(w[k * DF + n]);
}

__global__ void k_zero_stats(float* s) { s[threadIdx.x] = 0.0f; }

// ---------------- WMMA GEMM: H[N,64] = X[N,64] @ W[64,64] ----------------
// One wave per 16-row block. A fragments built with contiguous k (k = 16*half + s)
// (lane half selects K-bank); B fragments are pre-swizzled -> one v16u
// (32B, two b128 loads) per fragment per lane.

__global__ __launch_bounds__(256) void k_gemm_wmma(const float* __restrict__ X,
                                                   const v16u* __restrict__ Wswz,
                                                   float* __restrict__ H, int nrb) {
  const int wave = (blockIdx.x * blockDim.x + threadIdx.x) >> 5;
  if (wave >= nrb) return;
  const int lane = threadIdx.x & 31;
  const int nlo  = lane & 15;     // A-row within tile / C column / B column
  const int half = lane >> 4;

  // ---- B fragments: 8 contiguous 32B vector loads (coalesced) ----
  v16bf bfrag[8];
  #pragma unroll
  for (int f = 0; f < 8; ++f)
    bfrag[f] = __builtin_bit_cast(v16bf, Wswz[f * 32 + lane]);

  // ---- A fragments for K-steps 0 and 1 (K base 0 and 32) ----
  v16bf afrag[2];
  const int row = wave * 16 + nlo;
  #pragma unroll
  for (int ks = 0; ks < 2; ++ks) {
    const float* xr = X + (size_t)row * DF + ks * 32 + 16 * half;
    float4 c0 = *(const float4*)(xr + 0);
    float4 c1 = *(const float4*)(xr + 4);
    float4 c2 = *(const float4*)(xr + 8);
    float4 c3 = *(const float4*)(xr + 12);
    v16u u;
    u[0]  = f2bf_bits(c0.x); u[1]  = f2bf_bits(c0.y);
    u[2]  = f2bf_bits(c0.z); u[3]  = f2bf_bits(c0.w);
    u[4]  = f2bf_bits(c1.x); u[5]  = f2bf_bits(c1.y);
    u[6]  = f2bf_bits(c1.z); u[7]  = f2bf_bits(c1.w);
    u[8]  = f2bf_bits(c2.x); u[9]  = f2bf_bits(c2.y);
    u[10] = f2bf_bits(c2.z); u[11] = f2bf_bits(c2.w);
    u[12] = f2bf_bits(c3.x); u[13] = f2bf_bits(c3.y);
    u[14] = f2bf_bits(c3.z); u[15] = f2bf_bits(c3.w);
    afrag[ks] = __builtin_bit_cast(v16bf, u);
  }

  // ---- 4 output tiles of 16x16 across the 64 output columns ----
  #pragma unroll
  for (int nt = 0; nt < 4; ++nt) {
    v8f acc = {};
    #pragma unroll
    for (int ks = 0; ks < 2; ++ks) {
      acc = __builtin_amdgcn_wmma_f32_16x16x32_bf16(
          /*neg_a=*/false, afrag[ks], /*neg_b=*/false, bfrag[ks * 4 + nt],
          /*c_mod=*/(short)0, acc, /*reuse_a=*/false, /*reuse_b=*/false);
    }
    float* hp = H + (size_t)(wave * 16 + 8 * half) * DF + nt * 16 + nlo;
    #pragma unroll
    for (int r = 0; r < 8; ++r) hp[(size_t)r * DF] = acc[r];
  }
}

// ---------------- message passing ----------------

// agg[i,:] = dinv[i]^2 * h[i,:]   (self-loop term; also initializes agg)
__global__ __launch_bounds__(256) void k_self_init(const float* __restrict__ h,
                                                   const float* __restrict__ dinv,
                                                   float* __restrict__ agg, size_t tot) {
  size_t i = (size_t)blockIdx.x * blockDim.x + threadIdx.x;
  if (i >= tot) return;
  float di = dinv[i >> 6];
  agg[i] = di * di * h[i];
}

// One wave per edge: scalar-broadcast src/dst/norm, lanes cover the 64 dims
// with one float2 load + two fp32 global atomics each (L2-resident).
__global__ __launch_bounds__(256) void k_edge_scatter(const long long* __restrict__ ei,
                                                      const float* __restrict__ dinv,
                                                      const float* __restrict__ h,
                                                      float* __restrict__ agg, long long e) {
  long long t = (long long)blockIdx.x * blockDim.x + threadIdx.x;
  long long ed = t >> 5;
  if (ed >= e) return;
  int lane = (int)(t & 31);
  int s = (int)ei[ed];           // edge_index[0] = src
  int d = (int)ei[e + ed];       // edge_index[1] = dst
  float nrm = dinv[s] * dinv[d];
  float2 hv = *(const float2*)(h + (size_t)s * DF + lane * 2);
  float* ap = agg + (size_t)d * DF + lane * 2;
  atomicAdd(ap + 0, hv.x * nrm);
  atomicAdd(ap + 1, hv.y * nrm);
}

// ---------------- bias + ReLU + BN statistics (pass 1, in place) ----------------
__global__ __launch_bounds__(256) void k_bn_pass1(float* __restrict__ agg,
                                                  const float* __restrict__ bias,
                                                  float* __restrict__ stats, int nrows) {
  __shared__ float ls[DF], ls2[DF];
  const int col = threadIdx.x & 63;
  const int rg  = threadIdx.x >> 6;          // 4 row-threads per column
  if (threadIdx.x < DF) { ls[threadIdx.x] = 0.0f; ls2[threadIdx.x] = 0.0f; }
  __syncthreads();
  const int r0   = blockIdx.x * 256;
  const int rend = (r0 + 256 < nrows) ? (r0 + 256) : nrows;
  const float b = bias[col];
  float s = 0.0f, s2 = 0.0f;
  for (int r = r0 + rg; r < rend; r += 4) {
    size_t idx = (size_t)r * DF + col;
    float v = fmaxf(agg[idx] + b, 0.0f);
    agg[idx] = v;                             // y = relu(agg + b), in place
    s += v; s2 += v * v;
  }
  atomicAdd(&ls[col], s);
  atomicAdd(&ls2[col], s2);
  __syncthreads();
  if (threadIdx.x < DF) {
    atomicAdd(&stats[threadIdx.x],      ls[threadIdx.x]);
    atomicAdd(&stats[DF + threadIdx.x], ls2[threadIdx.x]);
  }
}

// ---------------- BN normalize (pass 2) ----------------
__global__ __launch_bounds__(256) void k_bn_pass2(const float* __restrict__ y,
                                                  const float* __restrict__ stats,
                                                  const float* __restrict__ gamma,
                                                  const float* __restrict__ beta,
                                                  float* __restrict__ out,
                                                  size_t tot, float invN) {
  size_t i = (size_t)blockIdx.x * blockDim.x + threadIdx.x;
  if (i >= tot) return;
  int col = (int)(i & 63);
  float mean = stats[col] * invN;
  float var  = stats[DF + col] * invN - mean * mean;
  out[i] = gamma[col] * (y[i] - mean) * rsqrtf(var + EPSF) + beta[col];
}

// ---------------- host orchestration ----------------

extern "C" void kernel_launch(void* const* d_in, const int* in_sizes, int n_in,
                              void* d_out, int out_size, void* d_ws, size_t ws_size,
                              hipStream_t stream) {
  (void)n_in; (void)out_size; (void)ws_size;

  const long long* ei   = (const long long*)d_in[0];
  const float*     node = (const float*)d_in[1];
  // d_in[2] = edge_attr (unused by the reference)
  const float* W[3]     = {(const float*)d_in[3],  (const float*)d_in[7],  (const float*)d_in[11]};
  const float* bias[3]  = {(const float*)d_in[4],  (const float*)d_in[8],  (const float*)d_in[12]};
  const float* gamma[3] = {(const float*)d_in[5],  (const float*)d_in[9],  (const float*)d_in[13]};
  const float* beta[3]  = {(const float*)d_in[6],  (const float*)d_in[10], (const float*)d_in[14]};

  const int       Nn  = in_sizes[1] / DF;            // 100000
  const long long Ee  = (long long)in_sizes[0] / 2;  // 1600000
  const size_t    tot = (size_t)Nn * DF;

  // workspace layout
  char* ws = (char*)d_ws;
  float* deg   = (float*)ws;                           // N
  float* dinv  = deg + Nn;                             // N
  float* stats = dinv + Nn;                            // 128 (sum, sumsq)
  // swizzled bf16 weights; keep 32B alignment for v16u loads
  size_t woff = ((size_t)((char*)(stats + 128) - ws) + 31) & ~(size_t)31;
  unsigned short* Wswz = (unsigned short*)(ws + woff); // 3 * 4096 bf16
  size_t off = (woff + 3 * 4096 * sizeof(unsigned short) + 255) & ~(size_t)255;
  float* P   = (float*)(ws + off);                     // ping node buffer [N,64]
  float* H   = P + tot;                                // x @ W            [N,64]
  float* AGG = H + tot;                                // aggregation / y  [N,64]

  const int nblkN   = (Nn + 255) / 256;
  const int nblkTot = (int)((tot + 255) / 256);
  const int nblkE   = (int)((Ee + 255) / 256);
  const int nblkSc  = (int)((Ee * 32 + 255) / 256);
  const int nrb     = Nn / 16;                         // 6250 row blocks (exact)

  // degrees (with self loop) and D^{-1/2}
  k_fill_deg <<<nblkN, 256, 0, stream>>>(deg, Nn);
  k_deg_accum<<<nblkE, 256, 0, stream>>>(ei, deg, Ee);
  k_dinv     <<<nblkN, 256, 0, stream>>>(deg, dinv, Nn);
  k_convert_w<<<48,    256, 0, stream>>>(W[0], W[1], W[2], Wswz);

  for (int l = 0; l < 3; ++l) {
    const float* xin = (l == 0) ? node : P;
    float* xout = (l == 2) ? (float*)d_out : P;

    k_zero_stats  <<<1, 128, 0, stream>>>(stats);
    k_gemm_wmma   <<<(nrb + 7) / 8, 256, 0, stream>>>(
        xin, (const v16u*)(Wswz + (size_t)l * 4096), H, nrb);
    k_self_init   <<<nblkTot, 256, 0, stream>>>(H, dinv, AGG, tot);
    k_edge_scatter<<<nblkSc, 256, 0, stream>>>(ei, dinv, H, AGG, Ee);
    k_bn_pass1    <<<nblkN, 256, 0, stream>>>(AGG, bias[l], stats, Nn);
    k_bn_pass2    <<<nblkTot, 256, 0, stream>>>(AGG, stats, gamma[l], beta[l],
                                                xout, tot, 1.0f / (float)Nn);
  }
}
